// SwinTransformer_1185410974386
// MI455X (gfx1250) — matrix operand order repaired and checked by reading, not compiled, not run on hardware
//
#include <hip/hip_runtime.h>
#include <hip/hip_bf16.h>
#include <math.h>

// ---------------- CDNA5 WMMA plumbing (wave32) ----------------
typedef __attribute__((ext_vector_type(16))) _Float16 v16h;
typedef __attribute__((ext_vector_type(8)))  float    v8f;

#define TOKS   100352   // 8 * 112 * 112
#define CDIM   256
#define NWIN   2048     // 8 * 16 * 16 windows
#define NPAD   64       // 49 tokens padded to 64
#define QKVC   768
#define NHEADS 8
#define HDIM   32

__device__ __forceinline__ v8f vzero8() {
  v8f z;
#pragma unroll
  for (int i = 0; i < 8; ++i) z[i] = 0.f;
  return z;
}

__device__ __forceinline__ v8f wmma_f16(v16h a, v16h b, v8f c) {
  // emits v_wmma_f32_16x16x32_f16
  return __builtin_amdgcn_wmma_f32_16x16x32_f16(false, a, false, b, (short)0, c,
                                                false, false);
}

// A fragment: 16x32 (MxK) f16, per ISA table: lanes 0-15 M=0..15 (dup 16-31),
// VGPR j<4: K = {0|8}+2j+half ; VGPR j>=4: K = {16|24}+2(j-4)+half
__device__ __forceinline__ v16h frag_a_gather(const _Float16* p, int sM, int sK) {
  const int lane = threadIdx.x & 31;
  const int m = lane & 15;
  const int kb0 = (lane < 16) ? 0 : 8;
  const int kb1 = (lane < 16) ? 16 : 24;
  v16h a;
#pragma unroll
  for (int h = 0; h < 16; ++h) {
    const int j = h >> 1, hs = h & 1;
    const int k = (j < 4) ? (kb0 + 2 * j + hs) : (kb1 + 2 * (j - 4) + hs);
    a[h] = p[m * sM + k * sK];
  }
  return a;
}

// B fragment: 32x16 (KxN) f16: lane holds N=lane%16; same K order as A:
// elem h<8 -> K={0|8}+h ; elem h>=8 -> K={16|24}+(h-8)
__device__ __forceinline__ v16h frag_b_gather(const _Float16* p, int sK, int sN) {
  const int lane = threadIdx.x & 31;
  const int n = lane & 15;
  const int kb = (lane < 16) ? 0 : 8;
  v16h b;
#pragma unroll
  for (int h = 0; h < 16; ++h) {
    b[h] = p[((h < 8) ? (kb + h) : (kb + 8 + h)) * sK + n * sN];
  }
  return b;
}

// ---------------- fp32 -> f16 weight conversion ----------------
__global__ void cvt_f32_f16_kernel(const float* __restrict__ src,
                                   _Float16* __restrict__ dst, int n) {
  int i = blockIdx.x * blockDim.x + threadIdx.x;
  if (i < n) dst[i] = (_Float16)src[i];
}

// ---------------- LayerNorm1 + shift + window partition -> f16 ----------------
// grid 12544 x block 256 (8 waves, one token per wave)
__global__ void ln_partition_kernel(const float* __restrict__ x,
                                    const float* __restrict__ g,
                                    const float* __restrict__ bta,
                                    _Float16* __restrict__ winx, int shift) {
  const int wave = threadIdx.x >> 5, lane = threadIdx.x & 31;
  const int id = blockIdx.x * 8 + wave;           // 0 .. 2048*49-1
  const int w = id / 49, t = id % 49;
  const int bi = w >> 8, rem = w & 255, wi = rem >> 4, wj = rem & 15;
  const int r = t / 7, c = t % 7;
  const int hi = (wi * 7 + r + shift) % 112;
  const int wq = (wj * 7 + c + shift) % 112;
  const long tok = (long)bi * 12544 + hi * 112 + wq;
  const float* xp = x + tok * CDIM;
  float v[8], s = 0.f, ss = 0.f;
#pragma unroll
  for (int j = 0; j < 8; ++j) {
    v[j] = xp[j * 32 + lane];
    s += v[j]; ss += v[j] * v[j];
  }
#pragma unroll
  for (int m = 16; m >= 1; m >>= 1) {
    s  += __shfl_xor(s,  m, 32);
    ss += __shfl_xor(ss, m, 32);
  }
  const float mean = s * (1.f / 256.f);
  const float rstd = rsqrtf(ss * (1.f / 256.f) - mean * mean + 1e-5f);
  _Float16* op = winx + (long)w * NPAD * CDIM + t * CDIM;
#pragma unroll
  for (int j = 0; j < 8; ++j) {
    const int ch = j * 32 + lane;
    op[ch] = (_Float16)((v[j] - mean) * rstd * g[ch] + bta[ch]);
  }
}

// ---------------- QKV GEMM: [64x256] @ [256x768]^T + bias (q scaled) ----------------
// grid 2048 x block 256, dyn LDS = 32768 (A) + 16384 (B slice) = 49152
// 3 column passes of 256 cols; each wave owns 32 cols -> acc[4][2] (64 VGPRs, no spill)
__global__ void qkv_gemm_kernel(const _Float16* __restrict__ winx,
                                const _Float16* __restrict__ wq,   // [768][256] f16
                                const float* __restrict__ bq,      // [768]
                                _Float16* __restrict__ qkv) {      // [2048][64][768]
  extern __shared__ char smem[];
  _Float16* lA = (_Float16*)smem;                   // [64][256]
  _Float16* lB = (_Float16*)(smem + 64 * 256 * 2);  // [256][32]
  const int tid = threadIdx.x, waveId = tid >> 5, lane = tid & 31;
  const _Float16* gA = winx + (long)blockIdx.x * NPAD * CDIM;
  for (int i = tid; i < 8192; i += 256)
    ((uint32_t*)lA)[i] = ((const uint32_t*)gA)[i];

  _Float16* out = qkv + (long)blockIdx.x * NPAD * QKVC;
  const int mb = (lane < 16) ? 0 : 8;
  const int n0 = waveId * 32;

  for (int pass = 0; pass < 3; ++pass) {
    v8f acc[4][2];
#pragma unroll
    for (int mt = 0; mt < 4; ++mt)
#pragma unroll
      for (int nt = 0; nt < 2; ++nt) acc[mt][nt] = vzero8();

    for (int kk = 0; kk < 8; ++kk) {
      __syncthreads();
      for (int i = tid; i < 8192; i += 256) {
        const int n = i >> 5, k = i & 31;
        lB[i] = wq[(pass * 256 + n) * CDIM + kk * 32 + k];
      }
      __syncthreads();
      v16h bf[2];
#pragma unroll
      for (int nt = 0; nt < 2; ++nt)
        bf[nt] = frag_b_gather(lB + (n0 + nt * 16) * 32, 1, 32);
#pragma unroll
      for (int mt = 0; mt < 4; ++mt) {
        v16h a = frag_a_gather(lA + mt * 16 * CDIM + kk * 32, CDIM, 1);
#pragma unroll
        for (int nt = 0; nt < 2; ++nt) acc[mt][nt] = wmma_f16(a, bf[nt], acc[mt][nt]);
      }
    }
    // epilogue: bias, q-scale, store f16 [t][col]
#pragma unroll
    for (int mt = 0; mt < 4; ++mt)
#pragma unroll
      for (int nt = 0; nt < 2; ++nt) {
        const int col = pass * 256 + n0 + nt * 16 + (lane & 15);
        const float bb = bq[col];
        const float scl = (col < 256) ? 0.17677669529663687f : 1.0f; // 1/sqrt(32) for q
#pragma unroll
        for (int r = 0; r < 8; ++r) {
          const int t = mt * 16 + mb + r;
          out[t * QKVC + col] = (_Float16)((acc[mt][nt][r] + bb) * scl);
        }
      }
  }
}

// ---------------- Attention per (window, head) ----------------
// grid 16384 x block 128 (4 waves), dyn LDS = 16384 (scores f32) + 8192 (P f16)
__global__ void attn_kernel(const _Float16* __restrict__ qkv,
                            const float* __restrict__ rpb,  // [169][8]
                            _Float16* __restrict__ aout,    // [2048][64][256]
                            int shift) {
  extern __shared__ char smem[];
  float* sc = (float*)smem;                         // [64][64]
  _Float16* P = (_Float16*)(smem + 64 * 64 * 4);    // [64][64]
  const int w = blockIdx.x >> 3, head = blockIdx.x & 7;
  const int tid = threadIdx.x, wave = tid >> 5, lane = tid & 31;
  const int rem = w & 255, wi = rem >> 4, wj = rem & 15;

  const _Float16* base = qkv + (long)w * NPAD * QKVC;
  const _Float16* qb = base + head * HDIM;
  const _Float16* kb = base + 256 + head * HDIM;
  const _Float16* vb = base + 512 + head * HDIM;
  const int mb = (lane < 16) ? 0 : 8;

  // scores = q @ k^T  (K = 32, single WMMA step per tile)
  {
    v16h a = frag_a_gather(qb + wave * 16 * QKVC, QKVC, 1);
#pragma unroll
    for (int nt = 0; nt < 4; ++nt) {
      v16h b = frag_b_gather(kb + nt * 16 * QKVC, 1, QKVC);
      v8f c = wmma_f16(a, b, vzero8());
      const int n = nt * 16 + (lane & 15);
#pragma unroll
      for (int r = 0; r < 8; ++r) {
        const int m = wave * 16 + mb + r;
        float val = c[r];
        if (m < 49 && n < 49) {
          const int r1 = m / 7, c1 = m % 7, r2 = n / 7, c2 = n % 7;
          const int idx = (r1 - r2 + 6) * 13 + (c1 - c2 + 6);
          val += rpb[idx * NHEADS + head];
          if (shift > 0) {
            const int gh1 = wi * 7 + r1, gw1 = wj * 7 + c1;
            const int gh2 = wi * 7 + r2, gw2 = wj * 7 + c2;
            const int lh1 = (gh1 < 105) ? 0 : ((gh1 < 109) ? 1 : 2);
            const int lw1 = (gw1 < 105) ? 0 : ((gw1 < 109) ? 1 : 2);
            const int lh2 = (gh2 < 105) ? 0 : ((gh2 < 109) ? 1 : 2);
            const int lw2 = (gw2 < 105) ? 0 : ((gw2 < 109) ? 1 : 2);
            if (lh1 * 3 + lw1 != lh2 * 3 + lw2) val += -100.0f;
          }
        }
        sc[m * 64 + n] = val;
      }
    }
  }
  __syncthreads();

  // fp32 softmax over the 49 valid keys; padded rows/cols -> P = 0
  if (tid < 64) {
    const int m = tid;
    if (m < 49) {
      float mx = -1e30f;
      for (int n = 0; n < 49; ++n) mx = fmaxf(mx, sc[m * 64 + n]);
      float s = 0.f;
      for (int n = 0; n < 49; ++n) {
        const float e = __expf(sc[m * 64 + n] - mx);
        s += e;
        sc[m * 64 + n] = e;
      }
      const float inv = 1.f / s;
      for (int n = 0; n < 64; ++n)
        P[m * 64 + n] = (n < 49) ? (_Float16)(sc[m * 64 + n] * inv) : (_Float16)0.f;
    } else {
      for (int n = 0; n < 64; ++n) P[m * 64 + n] = (_Float16)0.f;
    }
  }
  __syncthreads();

  // out = P @ V   (M=64, N=32, K=64)
  _Float16* ob = aout + (long)w * NPAD * CDIM + head * HDIM;
#pragma unroll
  for (int nt2 = 0; nt2 < 2; ++nt2) {
    v8f c = vzero8();
#pragma unroll
    for (int kk = 0; kk < 2; ++kk) {
      v16h a = frag_a_gather(P + wave * 16 * 64 + kk * 32, 64, 1);
      v16h b = frag_b_gather(vb + kk * 32 * QKVC + nt2 * 16, QKVC, 1);
      c = wmma_f16(a, b, c);
    }
    const int n = nt2 * 16 + (lane & 15);
#pragma unroll
    for (int r = 0; r < 8; ++r) {
      const int m = wave * 16 + mb + r;
      ob[m * CDIM + n] = (_Float16)c[r];
    }
  }
}

// ---------------- Proj GEMM + un-shift scatter + residual ----------------
// grid 2048 x block 256, dyn LDS = 32768 + 16384 = 49152
__global__ void proj_res_kernel(const _Float16* __restrict__ aout,
                                const _Float16* __restrict__ wp,  // [256][256] f16
                                const float* __restrict__ bp,
                                float* __restrict__ x, int shift) {
  extern __shared__ char smem[];
  _Float16* lA = (_Float16*)smem;                   // [64][256]
  _Float16* lB = (_Float16*)(smem + 64 * 256 * 2);  // [256][32]
  const int tid = threadIdx.x, waveId = tid >> 5, lane = tid & 31;
  const _Float16* gA = aout + (long)blockIdx.x * NPAD * CDIM;
  for (int i = tid; i < 8192; i += 256)
    ((uint32_t*)lA)[i] = ((const uint32_t*)gA)[i];

  const int n0 = waveId * 32;
  const int mb = (lane < 16) ? 0 : 8;
  v8f acc[4][2];
#pragma unroll
  for (int mt = 0; mt < 4; ++mt)
#pragma unroll
    for (int nt = 0; nt < 2; ++nt) acc[mt][nt] = vzero8();

  for (int kk = 0; kk < 8; ++kk) {
    __syncthreads();
    for (int i = tid; i < 8192; i += 256) {
      const int n = i >> 5, k = i & 31;
      lB[i] = wp[n * CDIM + kk * 32 + k];
    }
    __syncthreads();
    v16h bf[2];
#pragma unroll
    for (int nt = 0; nt < 2; ++nt)
      bf[nt] = frag_b_gather(lB + (n0 + nt * 16) * 32, 1, 32);
#pragma unroll
    for (int mt = 0; mt < 4; ++mt) {
      v16h a = frag_a_gather(lA + mt * 16 * CDIM + kk * 32, CDIM, 1);
#pragma unroll
      for (int nt = 0; nt < 2; ++nt) acc[mt][nt] = wmma_f16(a, bf[nt], acc[mt][nt]);
    }
  }

  const int w = blockIdx.x, bi = w >> 8, rem = w & 255, wi = rem >> 4, wj = rem & 15;
#pragma unroll
  for (int mt = 0; mt < 4; ++mt)
#pragma unroll
    for (int nt = 0; nt < 2; ++nt) {
      const int n = n0 + nt * 16 + (lane & 15);
      const float bb = bp[n];
#pragma unroll
      for (int r = 0; r < 8; ++r) {
        const int t = mt * 16 + mb + r;
        if (t < 49) {
          const int r1 = t / 7, c1 = t % 7;
          const int hi = (wi * 7 + r1 + shift) % 112;
          const int wq = (wj * 7 + c1 + shift) % 112;
          const long tok = (long)bi * 12544 + hi * 112 + wq;
          x[tok * CDIM + n] += acc[mt][nt][r] + bb;
        }
      }
    }
}

// ---------------- Fused LN2 + MLP (GEMM1 + GELU + GEMM2) + residual ----------------
// grid 1568 x block 256, dyn LDS = 32768*3 + 8192 = 106496
__global__ void mlp_kernel(float* __restrict__ x,
                           const float* __restrict__ g, const float* __restrict__ bta,
                           const _Float16* __restrict__ w1, const float* __restrict__ b1,
                           const _Float16* __restrict__ w2, const float* __restrict__ b2) {
  extern __shared__ char smem[];
  _Float16* lA  = (_Float16*)smem;                       // [64][256] LN2(x) f16
  _Float16* lW1 = (_Float16*)(smem + 32768);             // [64][256]
  _Float16* lW2 = (_Float16*)(smem + 65536);             // [256][64]
  _Float16* lH  = (_Float16*)(smem + 98304);             // [64][64]
  const int tid = threadIdx.x, waveId = tid >> 5, lane = tid & 31;
  const long tok0 = (long)blockIdx.x * 64;
  const int mb = (lane < 16) ? 0 : 8;

  // LN2 -> f16 A tile in LDS
  for (int rr = 0; rr < 8; ++rr) {
    const int row = waveId * 8 + rr;
    const float* xp = x + (tok0 + row) * CDIM;
    float v[8], s = 0.f, ss = 0.f;
#pragma unroll
    for (int j = 0; j < 8; ++j) {
      v[j] = xp[j * 32 + lane];
      s += v[j]; ss += v[j] * v[j];
    }
#pragma unroll
    for (int m = 16; m >= 1; m >>= 1) {
      s  += __shfl_xor(s,  m, 32);
      ss += __shfl_xor(ss, m, 32);
    }
    const float mean = s * (1.f / 256.f);
    const float rstd = rsqrtf(ss * (1.f / 256.f) - mean * mean + 1e-5f);
#pragma unroll
    for (int j = 0; j < 8; ++j) {
      const int ch = j * 32 + lane;
      lA[row * CDIM + ch] = (_Float16)((v[j] - mean) * rstd * g[ch] + bta[ch]);
    }
  }

  const int n0 = waveId * 32;
  v8f acc[4][2];
#pragma unroll
  for (int mt = 0; mt < 4; ++mt)
#pragma unroll
    for (int nt = 0; nt < 2; ++nt) acc[mt][nt] = vzero8();

  for (int ch = 0; ch < 16; ++ch) {
    __syncthreads();
    for (int i = tid; i < 16384; i += 256)
      lW1[i] = w1[ch * 64 * CDIM + i];                  // rows ch*64.. contiguous
    for (int i = tid; i < 16384; i += 256) {
      const int n = i >> 6, k = i & 63;
      lW2[i] = w2[n * 1024 + ch * 64 + k];
    }
    __syncthreads();

    // GEMM1: hidden chunk [64x64] = lA @ lW1^T ; bias + exact GELU -> lH f16
#pragma unroll
    for (int s2 = 0; s2 < 2; ++s2) {
      const int tileId = waveId * 2 + s2;
      const int mt = tileId >> 2, nt = tileId & 3;
      v8f h = vzero8();
      for (int kk = 0; kk < 8; ++kk) {
        v16h a = frag_a_gather(lA + mt * 16 * CDIM + kk * 32, CDIM, 1);
        v16h b = frag_b_gather(lW1 + nt * 16 * CDIM + kk * 32, 1, CDIM);
        h = wmma_f16(a, b, h);
      }
      const int n = nt * 16 + (lane & 15);
      const float bb = b1[ch * 64 + n];
#pragma unroll
      for (int r = 0; r < 8; ++r) {
        const int m = mt * 16 + mb + r;
        float v = h[r] + bb;
        v = 0.5f * v * (1.f + erff(v * 0.70710678118654752f));
        lH[m * 64 + n] = (_Float16)v;
      }
    }
    __syncthreads();

    // GEMM2: acc += gelu(h) @ lW2^T  (K = 64)
#pragma unroll
    for (int kk = 0; kk < 2; ++kk) {
      v16h bf[2];
#pragma unroll
      for (int nt2 = 0; nt2 < 2; ++nt2)
        bf[nt2] = frag_b_gather(lW2 + (n0 + nt2 * 16) * 64 + kk * 32, 1, 64);
#pragma unroll
      for (int mt = 0; mt < 4; ++mt) {
        v16h a = frag_a_gather(lH + mt * 16 * 64 + kk * 32, 64, 1);
#pragma unroll
        for (int nt2 = 0; nt2 < 2; ++nt2)
          acc[mt][nt2] = wmma_f16(a, bf[nt2], acc[mt][nt2]);
      }
    }
  }

  // residual + b2
#pragma unroll
  for (int mt = 0; mt < 4; ++mt)
#pragma unroll
    for (int nt2 = 0; nt2 < 2; ++nt2) {
      const int n = n0 + nt2 * 16 + (lane & 15);
      const float bb = b2[n];
#pragma unroll
      for (int r = 0; r < 8; ++r) {
        const int m = mt * 16 + mb + r;
        x[(tok0 + m) * CDIM + n] += acc[mt][nt2][r] + bb;
      }
    }
}

// ---------------- launcher ----------------
extern "C" void kernel_launch(void* const* d_in, const int* in_sizes, int n_in,
                              void* d_out, int out_size, void* d_ws, size_t ws_size,
                              hipStream_t stream) {
  (void)in_sizes; (void)n_in; (void)out_size; (void)ws_size;
  const float* x_in  = (const float*)d_in[0];
  const float* ln1w  = (const float*)d_in[1];
  const float* ln1b  = (const float*)d_in[2];
  const float* qkvw  = (const float*)d_in[3];
  const float* qkvb  = (const float*)d_in[4];
  const float* projw = (const float*)d_in[5];
  const float* projb = (const float*)d_in[6];
  const float* rpb   = (const float*)d_in[7];
  const float* ln2w  = (const float*)d_in[8];
  const float* ln2b  = (const float*)d_in[9];
  const float* w1    = (const float*)d_in[10];
  const float* b1    = (const float*)d_in[11];
  const float* w2    = (const float*)d_in[12];
  const float* b2    = (const float*)d_in[13];

  // workspace layout (~421 MiB total)
  char* ws = (char*)d_ws;
  size_t off = 0;
  auto carve = [&](size_t bytes) { char* p = ws + off; off = (off + bytes + 255) & ~(size_t)255; return p; };
  float*    X    = (float*)   carve((size_t)TOKS * CDIM * 4);         // 102.8 MB
  _Float16* WINX = (_Float16*)carve((size_t)NWIN * NPAD * CDIM * 2);  // 67.1 MB
  _Float16* QKV  = (_Float16*)carve((size_t)NWIN * NPAD * QKVC * 2);  // 201.3 MB
  _Float16* AOUT = (_Float16*)carve((size_t)NWIN * NPAD * CDIM * 2);  // 67.1 MB
  _Float16* WQ16 = (_Float16*)carve((size_t)2 * QKVC * CDIM * 2);
  _Float16* WP16 = (_Float16*)carve((size_t)2 * CDIM * CDIM * 2);
  _Float16* W116 = (_Float16*)carve((size_t)2 * 1024 * CDIM * 2);
  _Float16* W216 = (_Float16*)carve((size_t)2 * CDIM * 1024 * 2);

  // weights fp32 -> f16 (deterministic each call)
  auto cvt = [&](const float* s, _Float16* d, int n) {
    cvt_f32_f16_kernel<<<(n + 255) / 256, 256, 0, stream>>>(s, d, n);
  };
  cvt(qkvw,  WQ16, 2 * QKVC * CDIM);
  cvt(projw, WP16, 2 * CDIM * CDIM);
  cvt(w1,    W116, 2 * 1024 * CDIM);
  cvt(w2,    W216, 2 * CDIM * 1024);

  hipMemcpyAsync(X, x_in, (size_t)TOKS * CDIM * 4, hipMemcpyDeviceToDevice, stream);
  hipMemsetAsync(WINX, 0, (size_t)NWIN * NPAD * CDIM * 2, stream);  // zero padded rows

  for (int l = 0; l < 2; ++l) {
    const int shift = (l & 1) ? 3 : 0;
    ln_partition_kernel<<<TOKS / 8, 256, 0, stream>>>(
        X, ln1w + l * CDIM, ln1b + l * CDIM, WINX, shift);
    qkv_gemm_kernel<<<NWIN, 256, 49152, stream>>>(
        WINX, WQ16 + (size_t)l * QKVC * CDIM, qkvb + l * QKVC, QKV);
    attn_kernel<<<NWIN * NHEADS, 128, 24576, stream>>>(
        QKV, rpb + l * 169 * NHEADS, AOUT, shift);
    proj_res_kernel<<<NWIN, 256, 49152, stream>>>(
        AOUT, WP16 + (size_t)l * CDIM * CDIM, projb + l * CDIM, X, shift);
    mlp_kernel<<<TOKS / 64, 256, 106496, stream>>>(
        X, ln2w + l * CDIM, ln2b + l * CDIM,
        W116 + (size_t)l * 1024 * CDIM, b1 + l * 1024,
        W216 + (size_t)l * CDIM * 1024, b2 + l * CDIM);
  }

  hipMemcpyAsync(d_out, X, (size_t)TOKS * CDIM * 4, hipMemcpyDeviceToDevice, stream);
}
